// SegFormerBlock_2267742732938
// MI455X (gfx1250) — matrix operand order repaired and checked by reading, not compiled, not run on hardware
//
#include <hip/hip_runtime.h>
#include <hip/hip_fp16.h>

typedef __attribute__((ext_vector_type(16))) _Float16 v16h;
typedef __attribute__((ext_vector_type(8)))  float    v8f;

__device__ __forceinline__ int lane_id() { return threadIdx.x & 31; }

// K-index inside a 16-bit 16x32 A fragment (ISA 7.12.2):
// VGPR i<8 -> K=(lane>=16?8:0)+i ; i>=8 -> K=16+(lane>=16?8:0)+(i-8)
__device__ __forceinline__ int a_kidx(int lane, int i) {
  return ((i >> 3) << 4) + ((lane >> 4) << 3) + (i & 7);
}

// A fragment: rows m0..m0+15, cols k0..k0+31 of row-major [*,lda] f16 matrix
__device__ __forceinline__ v16h load_a(const _Float16* A, int lda, int m0, int k0) {
  const int lane = lane_id();
  const _Float16* row = A + (size_t)(m0 + (lane & 15)) * lda + k0;
  v16h a;
#pragma unroll
  for (int i = 0; i < 16; ++i) a[i] = row[a_kidx(lane, i)];
  return a;
}

// B fragment: rows k0..k0+31, cols n0..n0+15 of row-major [K,ldb] f16 matrix
// lane holds column n0+(lane&15), element i is K-row k0+a_kidx(lane,i)
__device__ __forceinline__ v16h load_b(const _Float16* B, int ldb, int k0, int n0) {
  const int lane = lane_id();
  const _Float16* row = B + (size_t)k0 * ldb + n0 + (lane & 15);
  v16h b;
#pragma unroll
  for (int i = 0; i < 16; ++i) b[i] = row[(size_t)a_kidx(lane, i) * ldb];
  return b;
}

__device__ __forceinline__ v8f wmma32(v16h a, v16h b, v8f c) {
  return __builtin_amdgcn_wmma_f32_16x16x32_f16(false, a, false, b, (short)0, c,
                                                false, false);
}

// Keep all fragment loads of one k-step clustered before the WMMA quartet so
// the waits are partial and WMMAs overlap outstanding loads.
__device__ __forceinline__ void sched_fence() { __builtin_amdgcn_sched_barrier(0); }

// ---------------------------------------------------------------- utilities

__global__ void cvt_f16_kernel(const float* __restrict__ src,
                               _Float16* __restrict__ dst, int n) {
  int i = blockIdx.x * blockDim.x + threadIdx.x;
  if (i < n) dst[i] = (_Float16)src[i];
}

// LayerNorm over last dim 64; one wave per row, 2 elems per lane. f32 in, f16 out.
__global__ void layernorm_kernel(const float* __restrict__ in,
                                 const float* __restrict__ g,
                                 const float* __restrict__ b,
                                 _Float16* __restrict__ out, int Mrows) {
  const int row = blockIdx.x * (blockDim.x >> 5) + (threadIdx.x >> 5);
  if (row >= Mrows) return;
  const int lane = lane_id();
  const float* p = in + (size_t)row * 64;
  float x0 = p[lane], x1 = p[lane + 32];
  float s = x0 + x1;
#pragma unroll
  for (int m = 16; m; m >>= 1) s += __shfl_xor(s, m, 32);
  float mean = s * (1.f / 64.f);
  float d0 = x0 - mean, d1 = x1 - mean;
  float v = d0 * d0 + d1 * d1;
#pragma unroll
  for (int m = 16; m; m >>= 1) v += __shfl_xor(v, m, 32);
  float rstd = rsqrtf(v * (1.f / 64.f) + 1e-5f);
  out[(size_t)row * 64 + lane]      = (_Float16)(d0 * rstd * g[lane] + b[lane]);
  out[(size_t)row * 64 + lane + 32] = (_Float16)(d1 * rstd * g[lane + 32] + b[lane + 32]);
}

// --------------------------------------------------- generic WMMA GEMM + epilogue
// C[M,N] = A[M,K] @ B[K,N] + bias (+ res). One wave computes a 16x64 strip
// (4 accumulator tiles): each A fragment feeds 4 WMMAs. Per k-step, A + 4 B
// fragments are issued as one load cluster (sched_fence) before the WMMAs.
template <bool OUT_F16, bool ADD_RES>
__global__ void gemm_wmma_kernel(const _Float16* __restrict__ A,
                                 const _Float16* __restrict__ Bm,
                                 const float* __restrict__ bias,
                                 const float* __restrict__ res,
                                 void* __restrict__ out, int M, int N, int K) {
  const int strips_n = N >> 6;
  const int strip = blockIdx.x * (blockDim.x >> 5) + (threadIdx.x >> 5);
  const int m0 = (strip / strips_n) << 4;
  const int nb = (strip % strips_n) << 6;
  if (m0 >= M) return;  // wave-uniform
  const int lane = lane_id();
  v8f acc[4] = {};
  for (int k0 = 0; k0 < K; k0 += 32) {
    v16h a = load_a(A, K, m0, k0);
    v16h b[4];
#pragma unroll
    for (int t = 0; t < 4; ++t) b[t] = load_b(Bm, N, k0, nb + 16 * t);
    sched_fence();
#pragma unroll
    for (int t = 0; t < 4; ++t) acc[t] = wmma32(a, b[t], acc[t]);
  }
#pragma unroll
  for (int t = 0; t < 4; ++t) {
    const int col = nb + 16 * t + (lane & 15);
    const float bv = bias ? bias[col] : 0.f;
#pragma unroll
    for (int r = 0; r < 8; ++r) {
      const int row = m0 + r + ((lane >> 4) << 3);
      float v = acc[t][r] + bv;
      if (ADD_RES) v += res[(size_t)row * N + col];
      if (OUT_F16) ((_Float16*)out)[(size_t)row * N + col] = (_Float16)v;
      else         ((float*)out)[(size_t)row * N + col] = v;
    }
  }
}

// ------------------------------------------- sr conv as 8x8-patchify GEMM
// out[2048,64] (f32, pre-LN) = patches(h)[2048,4096] @ sr_w[4096,64] + sr_b
// One wave per 16x64 strip: 128 strips total.
__global__ void sr_gemm_kernel(const _Float16* __restrict__ h16,
                               const _Float16* __restrict__ wsr,  // [4096,64]
                               const float* __restrict__ bias,
                               float* __restrict__ out) {
  const int strip = blockIdx.x * (blockDim.x >> 5) + (threadIdx.x >> 5);
  const int m0 = strip << 4;
  const int lane = lane_id();
  const int m = m0 + (lane & 15);               // patch index
  const int bb = m >> 8, p = m & 255, pr = p >> 4, pc = p & 15;
  const size_t base = ((size_t)bb * 16384 + (size_t)(pr * 8) * 128 + pc * 8) * 64;
  v8f acc[4] = {};
  for (int k0 = 0; k0 < 4096; k0 += 32) {
    v16h a;
#pragma unroll
    for (int i = 0; i < 16; ++i) {
      const int k = k0 + a_kidx(lane, i);       // k = ih*512 + j*64 + ci
      const int ih = k >> 9, j = (k >> 6) & 7, ci = k & 63;
      a[i] = h16[base + ((size_t)ih * 128 + j) * 64 + ci];
    }
    v16h b[4];
#pragma unroll
    for (int t = 0; t < 4; ++t) b[t] = load_b(wsr, 64, k0, 16 * t);
    sched_fence();
#pragma unroll
    for (int t = 0; t < 4; ++t) acc[t] = wmma32(a, b[t], acc[t]);
  }
#pragma unroll
  for (int t = 0; t < 4; ++t) {
    const int col = 16 * t + (lane & 15);
    const float bv = bias[col];
#pragma unroll
    for (int r = 0; r < 8; ++r)
      out[(size_t)(m0 + r + ((lane >> 4) << 3)) * 64 + col] = acc[t][r] + bv;
  }
}

// ------------------------------------------------------------- fused attention
// Per wave: 16 query rows. Scores (16x256) -> LDS, softmax in LDS, P@V via WMMA.
// 2 waves / block: 32KB f32 scores + 16KB f16 probs = 48KB LDS (of 320KB WGP).
__global__ void attention_kernel(const _Float16* __restrict__ q16,
                                 const _Float16* __restrict__ kv16,  // [B*256,128]
                                 _Float16* __restrict__ o16) {
  __shared__ float    S[2][16 * 256];
  __shared__ _Float16 P[2][16 * 256];
  const int wave = threadIdx.x >> 5;
  const int lane = lane_id();
  const int row0 = blockIdx.x * 32 + wave * 16;       // global query row
  const int batch = row0 >> 14;                        // /16384
  const _Float16* qb  = q16 + (size_t)row0 * 64;
  const _Float16* kvb = kv16 + (size_t)batch * 256 * 128;
  float* Sw = S[wave];
  _Float16* Pw = P[wave];

  // Pass 1: S = (Q @ K^T) * 1/sqrt(64); A fragments reused across all 16 n-tiles
  const v16h a0 = load_a(qb, 64, 0, 0);
  const v16h a1 = load_a(qb, 64, 0, 32);
  for (int n0 = 0; n0 < 256; n0 += 16) {
    v8f acc = {};
    v16h b0, b1;  // B=K^T: b[i] = K[n0+(lane&15)][k0+a_kidx(lane,i)], K row m = kvb[m*128 + 0..63]
#pragma unroll
    for (int i = 0; i < 16; ++i) {
      b0[i] = kvb[(size_t)(n0 + (lane & 15)) * 128 + a_kidx(lane, i)];
      b1[i] = kvb[(size_t)(n0 + (lane & 15)) * 128 + 32 + a_kidx(lane, i)];
    }
    sched_fence();
    acc = wmma32(a0, b0, acc);
    acc = wmma32(a1, b1, acc);
#pragma unroll
    for (int r = 0; r < 8; ++r)
      Sw[(r + ((lane >> 4) << 3)) * 256 + n0 + (lane & 15)] = acc[r] * 0.125f;
  }

  // Pass 2: softmax over 256 keys (wave-synchronous, no barrier needed)
  for (int r = 0; r < 16; ++r) {
    float mx = -3.4e38f;
    for (int c = lane; c < 256; c += 32) mx = fmaxf(mx, Sw[r * 256 + c]);
#pragma unroll
    for (int m = 16; m; m >>= 1) mx = fmaxf(mx, __shfl_xor(mx, m, 32));
    float sum = 0.f;
    for (int c = lane; c < 256; c += 32) sum += __expf(Sw[r * 256 + c] - mx);
#pragma unroll
    for (int m = 16; m; m >>= 1) sum += __shfl_xor(sum, m, 32);
    const float inv = 1.f / sum;
    for (int c = lane; c < 256; c += 32)
      Pw[r * 256 + c] = (_Float16)(__expf(Sw[r * 256 + c] - mx) * inv);
  }

  // Pass 3: O(16x64) = P(16x256) @ V(256x64); each P fragment feeds 4 WMMAs
  v8f acc[4] = {};
  for (int k0 = 0; k0 < 256; k0 += 32) {
    v16h a = load_a(Pw, 256, 0, k0);
    v16h b[4];
#pragma unroll
    for (int t = 0; t < 4; ++t) {
#pragma unroll
      for (int i = 0; i < 16; ++i)   // V = kvb cols 64..127
        b[t][i] = kvb[(size_t)(k0 + a_kidx(lane, i)) * 128 + 64 + 16 * t + (lane & 15)];
    }
    sched_fence();
#pragma unroll
    for (int t = 0; t < 4; ++t) acc[t] = wmma32(a, b[t], acc[t]);
  }
#pragma unroll
  for (int t = 0; t < 4; ++t)
#pragma unroll
    for (int r = 0; r < 8; ++r)
      o16[(size_t)(row0 + r + ((lane >> 4) << 3)) * 64 + 16 * t + (lane & 15)] =
          (_Float16)acc[t][r];
}

// ------------------------------------------- 3x3 depthwise conv (SAME) + exact GELU
__global__ void dwconv_gelu_kernel(const _Float16* __restrict__ f,  // [B,128,128,256]
                                   const float* __restrict__ w,     // [3,3,1,256]
                                   _Float16* __restrict__ g) {
  const size_t idx = (size_t)blockIdx.x * blockDim.x + threadIdx.x;
  const int c  = idx & 255;
  const int xx = (idx >> 8) & 127;
  const int yy = (idx >> 15) & 127;
  const int bb = (int)(idx >> 22);
  float acc = 0.f;
#pragma unroll
  for (int dy = -1; dy <= 1; ++dy) {
    const int y = yy + dy;
    if ((unsigned)y >= 128u) continue;
#pragma unroll
    for (int dx = -1; dx <= 1; ++dx) {
      const int x = xx + dx;
      if ((unsigned)x >= 128u) continue;
      acc += (float)f[(((size_t)bb * 128 + y) * 128 + x) * 256 + c] *
             w[((dy + 1) * 3 + (dx + 1)) * 256 + c];
    }
  }
  g[idx] = (_Float16)(0.5f * acc * (1.f + erff(acc * 0.70710678118f)));
}

// =================================================================== launch

extern "C" void kernel_launch(void* const* d_in, const int* in_sizes, int n_in,
                              void* d_out, int out_size, void* d_ws, size_t ws_size,
                              hipStream_t stream) {
  (void)in_sizes; (void)n_in; (void)out_size; (void)ws_size;
  const float* x      = (const float*)d_in[0];
  const float* ln1_g  = (const float*)d_in[3];
  const float* ln1_b  = (const float*)d_in[4];
  const float* q_w    = (const float*)d_in[5];
  const float* q_b    = (const float*)d_in[6];
  const float* kv_w   = (const float*)d_in[7];
  const float* kv_b   = (const float*)d_in[8];
  const float* proj_w = (const float*)d_in[9];
  const float* proj_b = (const float*)d_in[10];
  const float* sr_w   = (const float*)d_in[11];
  const float* sr_b   = (const float*)d_in[12];
  const float* srn_g  = (const float*)d_in[13];
  const float* srn_b  = (const float*)d_in[14];
  const float* ln2_g  = (const float*)d_in[15];
  const float* ln2_b  = (const float*)d_in[16];
  const float* fc1_w  = (const float*)d_in[17];
  const float* fc1_b  = (const float*)d_in[18];
  const float* dw_w   = (const float*)d_in[19];
  const float* fc2_w  = (const float*)d_in[20];
  const float* fc2_b  = (const float*)d_in[21];

  char* ws = (char*)d_ws;
  // f16 weight copies
  _Float16* wq16   = (_Float16*)(ws + 0);            // 4096
  _Float16* wkv16  = (_Float16*)(ws + 8192);         // 8192
  _Float16* wpj16  = (_Float16*)(ws + 24576);        // 4096
  _Float16* wsr16  = (_Float16*)(ws + 32768);        // 262144
  _Float16* wf1_16 = (_Float16*)(ws + 557056);       // 16384
  _Float16* wf2_16 = (_Float16*)(ws + 589824);       // 16384
  // activations
  _Float16* h16   = (_Float16*)(ws + (1u << 20));            // 16MB (reused as h2)
  _Float16* q16   = (_Float16*)(ws + 17825792u);             // 16MB
  _Float16* o16   = (_Float16*)(ws + 34603008u);             // 16MB
  _Float16* kv16  = (_Float16*)(ws + 51380224u);             // 512KB [2048,128]
  float*    xr32  = (float*)   (ws + 51904512u);             // 512KB [2048,64]
  _Float16* xrn16 = (_Float16*)(ws + 52428800u);             // 256KB
  float*    x1    = (float*)   (ws + 52690944u);             // 32MB  [131072,64]
  _Float16* fbuf  = (_Float16*)(ws + 86245376u);             // 64MB  [131072,256]
  _Float16* g16   = (_Float16*)(ws + 153354240u);            // 64MB

  // 0) weights -> f16
  cvt_f16_kernel<<<(4096 + 255) / 256, 256, 0, stream>>>(q_w, wq16, 4096);
  cvt_f16_kernel<<<(8192 + 255) / 256, 256, 0, stream>>>(kv_w, wkv16, 8192);
  cvt_f16_kernel<<<(4096 + 255) / 256, 256, 0, stream>>>(proj_w, wpj16, 4096);
  cvt_f16_kernel<<<(262144 + 255) / 256, 256, 0, stream>>>(sr_w, wsr16, 262144);
  cvt_f16_kernel<<<(16384 + 255) / 256, 256, 0, stream>>>(fc1_w, wf1_16, 16384);
  cvt_f16_kernel<<<(16384 + 255) / 256, 256, 0, stream>>>(fc2_w, wf2_16, 16384);

  // 1) h = LN1(x)                        [131072,64] f16
  layernorm_kernel<<<16384, 256, 0, stream>>>(x, ln1_g, ln1_b, h16, 131072);
  // 2) xr = patchify(h) @ sr_w + sr_b    [2048,64] f32   (128 strips / 8 waves)
  sr_gemm_kernel<<<16, 256, 0, stream>>>(h16, wsr16, sr_b, xr32);
  // 3) xrn = LN(xr)
  layernorm_kernel<<<256, 256, 0, stream>>>(xr32, srn_g, srn_b, xrn16, 2048);
  // 4) kv = xrn @ kv_w + kv_b            [2048,128] f16  (256 strips)
  gemm_wmma_kernel<true, false><<<32, 256, 0, stream>>>(
      xrn16, wkv16, kv_b, nullptr, kv16, 2048, 128, 64);
  // 5) q = h @ q_w + q_b                 [131072,64] f16 (8192 strips)
  gemm_wmma_kernel<true, false><<<1024, 256, 0, stream>>>(
      h16, wq16, q_b, nullptr, q16, 131072, 64, 64);
  // 6) o = softmax(q k^T / 8) v          [131072,64] f16
  attention_kernel<<<4096, 64, 0, stream>>>(q16, kv16, o16);
  // 7) x1 = x + o @ proj_w + proj_b      [131072,64] f32
  gemm_wmma_kernel<false, true><<<1024, 256, 0, stream>>>(
      o16, wpj16, proj_b, x, x1, 131072, 64, 64);
  // 8) h2 = LN2(x1)  (reuse h16)
  layernorm_kernel<<<16384, 256, 0, stream>>>(x1, ln2_g, ln2_b, h16, 131072);
  // 9) f = h2 @ fc1_w + fc1_b            [131072,256] f16 (32768 strips)
  gemm_wmma_kernel<true, false><<<4096, 256, 0, stream>>>(
      h16, wf1_16, fc1_b, nullptr, fbuf, 131072, 256, 64);
  // 10) g = gelu(dwconv3x3(f))           [131072,256] f16
  dwconv_gelu_kernel<<<131072, 256, 0, stream>>>(fbuf, dw_w, g16);
  // 11) out = x1 + g @ fc2_w + fc2_b     [131072,64] f32
  gemm_wmma_kernel<false, true><<<1024, 256, 0, stream>>>(
      g16, wf2_16, fc2_b, x1, (float*)d_out, 131072, 64, 256);
}
